// CortexIIBlock_54305566490741
// MI455X (gfx1250) — compile-verified, matrix operand order checked by reading, not run on hardware
//
#include <hip/hip_runtime.h>

// ---------------------------------------------------------------------------
// CortexII block for MI455X (gfx1250, wave32, WMMA).
// GEMMs: v_wmma_f32_16x16x32_bf16, double-buffered LDS, async global->LDS
// staging (ASYNCcnt path) when the toolchain exposes it.
// ---------------------------------------------------------------------------

typedef __bf16 bf16;
typedef __attribute__((ext_vector_type(4)))  __bf16 v4bf;
typedef __attribute__((ext_vector_type(8)))  __bf16 v8bf;
typedef __attribute__((ext_vector_type(16))) __bf16 v16bf;
typedef __attribute__((ext_vector_type(8)))  float  v8f;
typedef __attribute__((ext_vector_type(4)))  int    v4i;

#define LDA 40  // padded LDS row stride in bf16 elements (80B: conflict-free)

#if defined(__has_builtin)
#if __has_builtin(__builtin_amdgcn_global_load_async_to_lds_b128)
#define USE_ASYNC_LDS 1
#endif
#if __has_builtin(__builtin_amdgcn_s_wait_asynccnt)
#define HAVE_WAIT_ASYNC 1
#endif
#endif

// One 16-byte global->LDS copy (async DMA path if available).
static __device__ __forceinline__ void cp_b128(bf16* ldsDst, const bf16* gSrc) {
#if defined(USE_ASYNC_LDS)
  __builtin_amdgcn_global_load_async_to_lds_b128(
      (__attribute__((address_space(1))) v4i*)gSrc,
      (__attribute__((address_space(3))) v4i*)ldsDst, 0, 0);
#else
  *(uint4*)ldsDst = *(const uint4*)gSrc;
#endif
}

static __device__ __forceinline__ void wait_stage() {
#if defined(USE_ASYNC_LDS)
#if defined(HAVE_WAIT_ASYNC)
  __builtin_amdgcn_s_wait_asynccnt(0);
#else
  asm volatile("s_wait_asynccnt 0x0" ::: "memory");
#endif
#endif
}

static __device__ __forceinline__ float fast_sigmoid(float x) {
  return __builtin_amdgcn_rcpf(1.f + __expf(-x));  // v_rcp_f32, no IEEE div
}

// Load one 16x32 bf16 WMMA operand fragment from a row-major LDS tile.
// ISA layout (CDNA5 ISA): lanes 0-15 hold row (lane&15), K {0..7,16..23};
// lanes 16-31 hold same rows, K {8..15,24..31}.
static __device__ __forceinline__ v16bf load_frag(const bf16* base, int lane) {
  const int row = lane & 15;
  const int k0  = (lane & 16) ? 8 : 0;
  union { v16bf v; v8bf h[2]; } u;
  u.h[0] = *(const v8bf*)(base + row * LDA + k0);
  u.h[1] = *(const v8bf*)(base + row * LDA + k0 + 16);
  return u.v;
}

enum { EPI_GATEVAL = 0, EPI_RESADD = 1, EPI_SWIGLU = 2 };

// C = A(MxK bf16) * B(NxK bf16)^T, fused epilogues, double-buffered LDS.
// Block tile 128 x BN (BN=128 single-B, 64 dual-B); 8 waves as 4M x 2N;
// wave tile 32 x BN/2 -> 2 x TN WMMA tiles per wave per 32-K chunk.
template<int EPI, bool DUAL>
__global__ __launch_bounds__(256) void gemm_bf16_wmma(
    const bf16* __restrict__ A,  const bf16* __restrict__ B0,
    const bf16* __restrict__ B1, const float* __restrict__ Cin,
    float* __restrict__ outF, bf16* __restrict__ out0, bf16* __restrict__ out1,
    int M, int N, int K, int Nsplit)
{
  constexpr int BN = DUAL ? 64 : 128;
  constexpr int TN = BN / 32;            // N tiles per wave (2 or 4)
  constexpr int ASZ = 128 * LDA;
  constexpr int BSZ = BN * LDA;

  __shared__ bf16 As [2 * ASZ];
  __shared__ bf16 Bs0[2 * BSZ];
  __shared__ bf16 Bs1[DUAL ? 2 * BSZ : 1];

  const int tid  = threadIdx.x;
  const int bm   = blockIdx.y * 128;
  const int bn   = blockIdx.x * BN;
  const int wid  = tid >> 5;
  const int lane = tid & 31;
  const int wm   = (wid & 3) * 32;          // wave row offset
  const int wn   = (wid >> 2) * (BN / 2);   // wave col offset

  v8f acc [2][TN] = {};
  v8f accu[2][DUAL ? TN : 1] = {};

  // Stage one 128x32 A tile + BNx32 B tile(s) into buffer `buf`.
  auto stage = [&](int buf, int k0) {
#pragma unroll
    for (int i = 0; i < 2; ++i) {           // A: 512 b128 segs / 256 thr
      const int seg = tid + i * 256;
      const int row = seg >> 2, col = (seg & 3) * 8;
      cp_b128(&As[buf * ASZ + row * LDA + col],
              A + (long)(bm + row) * K + k0 + col);
    }
#pragma unroll
    for (int i = 0; i < (DUAL ? 1 : 2); ++i) {  // B0: BN*4 segs
      const int seg = tid + i * 256;
      const int row = seg >> 2, col = (seg & 3) * 8;
      cp_b128(&Bs0[buf * BSZ + row * LDA + col],
              B0 + (long)(bn + row) * K + k0 + col);
    }
    if (DUAL) {
      const int row = tid >> 2, col = (tid & 3) * 8;
      cp_b128(&Bs1[buf * BSZ + row * LDA + col],
              B1 + (long)(bn + row) * K + k0 + col);
    }
  };

  const int nChunks = K >> 5;
  stage(0, 0);
  for (int kc = 0; kc < nChunks; ++kc) {
    const int buf = kc & 1;
    wait_stage();          // my async loads into buf are complete
    __syncthreads();       // everyone's are (also guards buf reuse WAR)
    if (kc + 1 < nChunks) stage(buf ^ 1, (kc + 1) * 32);
#if !defined(USE_ASYNC_LDS)
    if (kc + 2 < nChunks) {
      __builtin_prefetch(A + (long)(bm + (tid >> 2)) * K + (kc + 2) * 32, 0, 1);
      __builtin_prefetch(B0 + (long)(bn + (tid >> 2) % BN) * K + (kc + 2) * 32, 0, 1);
    }
#endif
    const bf16* Ab  = &As [buf * ASZ];
    const bf16* B0b = &Bs0[buf * BSZ];
    v16bf af[2];
    af[0] = load_frag(Ab + (wm +  0) * LDA, lane);
    af[1] = load_frag(Ab + (wm + 16) * LDA, lane);
#pragma unroll
    for (int tn = 0; tn < TN; ++tn) {
      const v16bf bfrag0 = load_frag(B0b + (wn + tn * 16) * LDA, lane);
#pragma unroll
      for (int tm = 0; tm < 2; ++tm)
        acc[tm][tn] = __builtin_amdgcn_wmma_f32_16x16x32_bf16(
            false, af[tm], false, bfrag0, (short)0, acc[tm][tn], false, false);
      if (DUAL) {
        const v16bf bfrag1 = load_frag(&Bs1[buf * BSZ] + (wn + tn * 16) * LDA, lane);
#pragma unroll
        for (int tm = 0; tm < 2; ++tm)
          accu[tm][tn] = __builtin_amdgcn_wmma_f32_16x16x32_bf16(
              false, af[tm], false, bfrag1, (short)0, accu[tm][tn], false, false);
      }
    }
  }

  // Epilogue. C/D layout: VGPR r, lanes 0-15 -> M=r, N=lane; 16-31 -> M=8+r.
  const int nIn  = lane & 15;
  const int mOff = (lane >> 4) * 8;
#pragma unroll
  for (int tm = 0; tm < 2; ++tm)
#pragma unroll
    for (int tn = 0; tn < TN; ++tn)
#pragma unroll
      for (int r = 0; r < 8; ++r) {
        const int m = bm + wm + tm * 16 + mOff + r;
        const int n = bn + wn + tn * 16 + nIn;
        const float v = acc[tm][tn][r];
        if (EPI == EPI_GATEVAL) {
          if (n < Nsplit) out0[(long)m * Nsplit + n] = (bf16)fast_sigmoid(v);
          else            out1[(long)m * Nsplit + (n - Nsplit)] = (bf16)v;
        } else if (EPI == EPI_RESADD) {
          const long i = (long)m * N + n;
          outF[i] = Cin[i] + v;   // also covers in-place accumulate (Cin==outF)
        } else {                  // EPI_SWIGLU: silu(g) * u
          const float u = accu[tm][tn][r];
          out0[(long)m * N + n] = (bf16)(v * fast_sigmoid(v) * u);
        }
      }
}

// RMSNorm (one token per block), optionally fused 3-way scale-gate softmax.
template<bool SG>
__global__ __launch_bounds__(256) void rmsnorm_kernel(
    const float* __restrict__ x, const float* __restrict__ w,
    const float* __restrict__ sg_w, bf16* __restrict__ h,
    float* __restrict__ scale)
{
  constexpr int D = 1024;
  __shared__ float red[256];
  const int m = blockIdx.x;
  const int tid = threadIdx.x;
  const float* xr = x + (long)m * D;
  float v[4]; float ss = 0.f;
#pragma unroll
  for (int i = 0; i < 4; ++i) { v[i] = xr[tid + i * 256]; ss += v[i] * v[i]; }
  red[tid] = ss; __syncthreads();
  for (int s = 128; s > 0; s >>= 1) { if (tid < s) red[tid] += red[tid + s]; __syncthreads(); }
  const float r = rsqrtf(red[0] * (1.f / D) + 1e-6f);
  __syncthreads();
  float hv[4];
#pragma unroll
  for (int i = 0; i < 4; ++i) {
    const int d = tid + i * 256;
    hv[i] = v[i] * r * w[d];
    h[(long)m * D + d] = (bf16)hv[i];
  }
  if (SG) {
    float p[3] = {0.f, 0.f, 0.f};
#pragma unroll
    for (int i = 0; i < 4; ++i) {
      const int d = tid + i * 256;
      p[0] += hv[i] * sg_w[d];
      p[1] += hv[i] * sg_w[D + d];
      p[2] += hv[i] * sg_w[2 * D + d];
    }
    float l[3];
    for (int e = 0; e < 3; ++e) {
      red[tid] = p[e]; __syncthreads();
      for (int s = 128; s > 0; s >>= 1) { if (tid < s) red[tid] += red[tid + s]; __syncthreads(); }
      l[e] = red[0]; __syncthreads();
    }
    if (tid == 0) {
      const float mx = fmaxf(l[0], fmaxf(l[1], l[2]));
      const float e0 = __expf(l[0] - mx), e1 = __expf(l[1] - mx), e2 = __expf(l[2] - mx);
      const float inv = __builtin_amdgcn_rcpf(e0 + e1 + e2);
      scale[m * 3 + 0] = e0 * inv;
      scale[m * 3 + 1] = e1 * inv;
      scale[m * 3 + 2] = e2 * inv;
    }
  }
}

// Causal depthwise 3/5/7-tap convs, softmax-scale mix, sigmoid-gate product.
__global__ __launch_bounds__(256) void conv_fuse_kernel(
    const bf16* __restrict__ val, const bf16* __restrict__ gate,
    const float* __restrict__ scale, const float* __restrict__ wf,
    const float* __restrict__ wmed, const float* __restrict__ wc,
    bf16* __restrict__ y)
{
  constexpr int D = 1024, S = 4096;
  const long idx = (long)blockIdx.x * 256 + threadIdx.x;
  const int d = (int)(idx & (D - 1));
  const int m = (int)(idx >> 10);
  const int s = m & (S - 1);
  float v[7];
#pragma unroll
  for (int j = 0; j < 7; ++j) {
    const int so = s - 6 + j;  // causal zero pad, never crosses a sequence
    v[j] = (so >= 0) ? (float)val[(long)(m - 6 + j) * D + d] : 0.f;
  }
  float cf = 0.f, cm = 0.f, cc = 0.f;
#pragma unroll
  for (int k = 0; k < 3; ++k) cf += wf[d * 3 + k] * v[4 + k];
#pragma unroll
  for (int k = 0; k < 5; ++k) cm += wmed[d * 5 + k] * v[2 + k];
#pragma unroll
  for (int k = 0; k < 7; ++k) cc += wc[d * 7 + k] * v[k];
  const float s0 = scale[m * 3], s1 = scale[m * 3 + 1], s2 = scale[m * 3 + 2];
  const float g = (float)gate[idx];
  y[idx] = (bf16)((s0 * cf + s1 * cm + s2 * cc) * g);
}

// fp32 -> bf16 weight conversion (x4 vectorized; all counts are multiples of 4).
__global__ __launch_bounds__(256) void cvt_f32_bf16(
    const float4* __restrict__ in, v4bf* __restrict__ out, int n4)
{
  const int i = blockIdx.x * 256 + threadIdx.x;
  if (i < n4) {
    const float4 f = in[i];
    v4bf o;
    o[0] = (bf16)f.x; o[1] = (bf16)f.y; o[2] = (bf16)f.z; o[3] = (bf16)f.w;
    out[i] = o;
  }
}

extern "C" void kernel_launch(void* const* d_in, const int* in_sizes, int n_in,
                              void* d_out, int out_size, void* d_ws, size_t ws_size,
                              hipStream_t stream)
{
  (void)in_sizes; (void)n_in; (void)out_size; (void)ws_size;
  constexpr int Dm = 1024, S = 4096, Bt = 4, FF = 4096;
  constexpr int M  = Bt * S;  // 16384 tokens

  const float* x      = (const float*)d_in[0];
  const float* ln1_w  = (const float*)d_in[1];
  const float* ln2_w  = (const float*)d_in[2];
  const float* w_fine = (const float*)d_in[3];
  const float* w_med  = (const float*)d_in[4];
  const float* w_coa  = (const float*)d_in[5];
  const float* sg_w   = (const float*)d_in[6];
  const float* up_w   = (const float*)d_in[7];
  const float* down_w = (const float*)d_in[8];
  const float* wg     = (const float*)d_in[9];
  const float* wu     = (const float*)d_in[10];
  const float* wo     = (const float*)d_in[11];
  float* out = (float*)d_out;

  // Workspace layout (~190 MB). Region A (4x32MB: h,gate,val,y) is reused as
  // the 128MB g*u tensor once those buffers are dead.
  char* ws = (char*)d_ws;
  const size_t SZ_MD = (size_t)M * Dm * sizeof(bf16);  // 32 MB
  bf16* h    = (bf16*)(ws + 0 * SZ_MD);
  bf16* gate = (bf16*)(ws + 1 * SZ_MD);
  bf16* val  = (bf16*)(ws + 2 * SZ_MD);
  bf16* y    = (bf16*)(ws + 3 * SZ_MD);
  bf16* t    = (bf16*)(ws + 0 * SZ_MD);  // alias: [M x FF] bf16 = 128 MB
  bf16* h2   = (bf16*)(ws + 4 * SZ_MD);
  float* scale = (float*)(ws + 5 * SZ_MD);
  size_t woff = 5 * SZ_MD + (((size_t)M * 3 * sizeof(float) + 255) & ~(size_t)255);
  bf16* up_b   = (bf16*)(ws + woff); woff += (size_t)2 * Dm * Dm * sizeof(bf16);
  bf16* down_b = (bf16*)(ws + woff); woff += (size_t)Dm * Dm * sizeof(bf16);
  bf16* wg_b   = (bf16*)(ws + woff); woff += (size_t)FF * Dm * sizeof(bf16);
  bf16* wu_b   = (bf16*)(ws + woff); woff += (size_t)FF * Dm * sizeof(bf16);
  bf16* wo_b   = (bf16*)(ws + woff);

  auto cvt = [&](const float* src, bf16* dst, int n) {
    const int n4 = n / 4;
    cvt_f32_bf16<<<dim3((n4 + 255) / 256), dim3(256), 0, stream>>>(
        (const float4*)src, (v4bf*)dst, n4);
  };
  cvt(up_w,   up_b,   2 * Dm * Dm);
  cvt(down_w, down_b, Dm * Dm);
  cvt(wg,     wg_b,   FF * Dm);
  cvt(wu,     wu_b,   FF * Dm);
  cvt(wo,     wo_b,   Dm * FF);

  // 1) h = rmsnorm(x, ln1); scale = softmax(h @ sg_w^T)
  rmsnorm_kernel<true><<<dim3(M), dim3(256), 0, stream>>>(x, ln1_w, sg_w, h, scale);

  // 2) [gate|val] = h @ up_w^T (sigmoid fused on gate half)
  gemm_bf16_wmma<EPI_GATEVAL, false><<<dim3(2 * Dm / 128, M / 128), dim3(256), 0, stream>>>(
      h, up_b, nullptr, nullptr, nullptr, gate, val, M, 2 * Dm, Dm, Dm);

  // 3) y = (s0*c3 + s1*c5 + s2*c7)(val) * gate
  conv_fuse_kernel<<<dim3((unsigned)((long)M * Dm / 256)), dim3(256), 0, stream>>>(
      val, gate, scale, w_fine, w_med, w_coa, y);

  // 4) out = x + y @ down_w^T
  gemm_bf16_wmma<EPI_RESADD, false><<<dim3(Dm / 128, M / 128), dim3(256), 0, stream>>>(
      y, down_b, nullptr, x, out, nullptr, nullptr, M, Dm, Dm, 0);

  // 5) h2 = rmsnorm(out, ln2)
  rmsnorm_kernel<false><<<dim3(M), dim3(256), 0, stream>>>(out, ln2_w, nullptr, h2, nullptr);

  // 6) t = silu(h2 @ wg^T) * (h2 @ wu^T)   (dual-B, fused SwiGLU)
  gemm_bf16_wmma<EPI_SWIGLU, true><<<dim3(FF / 64, M / 128), dim3(256), 0, stream>>>(
      h2, wg_b, wu_b, nullptr, nullptr, t, nullptr, M, FF, Dm, 0);

  // 7) out += t @ wo^T
  gemm_bf16_wmma<EPI_RESADD, false><<<dim3(Dm / 128, M / 128), dim3(256), 0, stream>>>(
      t, wo_b, nullptr, out, out, nullptr, nullptr, M, Dm, FF, 0);
}
